// AttentionModel_24756191494742
// MI455X (gfx1250) — compile-verified, matrix operand order checked by reading, not run on hardware
//
#include <hip/hip_runtime.h>
#include <hip/hip_bf16.h>
#include <math.h>

typedef __bf16 bf16_t;
typedef __attribute__((ext_vector_type(16))) __bf16 v16bf;
typedef __attribute__((ext_vector_type(8)))  __bf16 v8bf;
typedef __attribute__((ext_vector_type(8)))  float  v8f;

#define DEV __device__ __forceinline__

DEV bf16_t to_bf16(float f) { return (bf16_t)f; }

// ---- gfx1250 async global->LDS (probe-confirmed available; takes v4i*) ------
#if __has_builtin(__builtin_amdgcn_global_load_async_to_lds_b128)
#define USE_ASYNC_LDS 1
#else
#define USE_ASYNC_LDS 0
#endif

typedef int v4i_b128 __attribute__((vector_size(16)));
#define GPTR(p) ((__attribute__((address_space(1))) v4i_b128*)(p))
#define LPTR(p) ((__attribute__((address_space(3))) v4i_b128*)(p))

DEV void wait_async0() {
#if __has_builtin(__builtin_amdgcn_s_wait_asynccnt)
    __builtin_amdgcn_s_wait_asynccnt(0);
#else
    asm volatile("s_wait_asynccnt 0x0" ::: "memory");
#endif
}

// ---------------- problem sizes (fixed by the reference) ----------------
constexpr int S_ = 2048, B_ = 4, D_ = 1024, A_ = 1024, F_ = 4096, O_ = 1024;

// ---------------- GEMM tiling ----------------
constexpr int BM = 128, BN = 128, BK = 32;
constexpr int PADK = 8, LDK = BK + PADK;   // 40 halves = 80B row stride (16B aligned)

enum { EPI_F32 = 0, EPI_BF16 = 1, EPI_GELU = 2, EPI_SCORES = 3 };

// ---------------------------------------------------------------------------
// Tiled bf16 WMMA GEMM:   C[M,N] = A[M,K] * B   with B given row-major [N,K]
// 256 threads = 8 waves; wave grid 2(M) x 4(N); wave tile 64x32 = 4x2 frags.
// LDS double-buffered; next K-tile staged ahead via async global->LDS copies
// (ASYNCcnt) when available, else via VGPR staging.
// ---------------------------------------------------------------------------
template <int EPI, bool HAS_BIAS, bool CAUSAL_SKIP, bool KLIMIT, bool OUT_TRANS>
__global__ __launch_bounds__(256)
void gemm_wmma(const bf16_t* __restrict__ A, int lda,
               const bf16_t* __restrict__ Bt, int ldb,   // row-major [N,K]
               void* __restrict__ Out, int ldOut,
               int Kdim,
               const float* __restrict__ bias,
               float scale,
               long aBS, long bBS, long oBS)
{
    __shared__ bf16_t As[2][BM][LDK];   // M-major:  As[buf][m][k]
    __shared__ bf16_t Bs[2][BN][LDK];   // N-major:  Bs[buf][n][k]

    const int bmI = blockIdx.y, bnI = blockIdx.x, bb = blockIdx.z;
    const int m0 = bmI * BM, n0 = bnI * BN;
    if (CAUSAL_SKIP) {
        if (n0 > m0 + BM - 1) return;   // tile entirely above causal diagonal
    }

    const bf16_t* Ab = A  + (long)bb * aBS;
    const bf16_t* Bb = Bt + (long)bb * bBS;

    int kEnd = Kdim;
    if (KLIMIT) kEnd = min(Kdim, m0 + BM);   // probs are zero past the diagonal
    const int kTiles = kEnd / BK;            // all K dims are multiples of 32

    const int tid  = threadIdx.x;
    const int lane = tid & 31;
    const int wave = tid >> 5;               // 0..7
    const int wm   = wave >> 2;              // 0..1  (M)
    const int wn   = wave & 3;               // 0..3  (N)

    // ---- staging: 2 threads per row, 32B (two b128 chunks) per side each
    const int rowS = tid >> 1, halfS = (tid & 1) * 16;
    const bf16_t* aSrc = Ab + (long)(m0 + rowS) * lda + halfS;
    const bf16_t* bSrc = Bb + (long)(n0 + rowS) * ldb + halfS;

#if USE_ASYNC_LDS
    // fire-and-forget global->LDS (tracked by ASYNCcnt); imm offset applies to
    // both the global and LDS address per the ISA.
    auto stage_async = [&](int k0, int buf) {
        bf16_t* la = &As[buf][rowS][halfS];
        bf16_t* lb = &Bs[buf][rowS][halfS];
        __builtin_amdgcn_global_load_async_to_lds_b128(GPTR(aSrc + k0), LPTR(la), 0,  0);
        __builtin_amdgcn_global_load_async_to_lds_b128(GPTR(aSrc + k0), LPTR(la), 16, 0);
        __builtin_amdgcn_global_load_async_to_lds_b128(GPTR(bSrc + k0), LPTR(lb), 0,  0);
        __builtin_amdgcn_global_load_async_to_lds_b128(GPTR(bSrc + k0), LPTR(lb), 16, 0);
    };
#else
    v8bf ra0, ra1, rb0, rb1;
    auto ldreg = [&](int k0) {
        ra0 = *(const v8bf*)(aSrc + k0);
        ra1 = *(const v8bf*)(aSrc + k0 + 8);
        rb0 = *(const v8bf*)(bSrc + k0);
        rb1 = *(const v8bf*)(bSrc + k0 + 8);
    };
    auto streg = [&](int buf) {
        *(v8bf*)&As[buf][rowS][halfS]     = ra0;
        *(v8bf*)&As[buf][rowS][halfS + 8] = ra1;
        *(v8bf*)&Bs[buf][rowS][halfS]     = rb0;
        *(v8bf*)&Bs[buf][rowS][halfS + 8] = rb1;
    };
#endif

    v8f acc[4][2] = {};

#if USE_ASYNC_LDS
    stage_async(0, 0);
    wait_async0();
    __syncthreads();
#else
    ldreg(0);
    streg(0);
    __syncthreads();
#endif

    const int r  = lane & 15;
    const int kh = (lane >> 4) * 8;

    for (int kt = 0; kt < kTiles; ++kt) {
        const int  cur  = kt & 1;
        const bool more = (kt + 1 < kTiles);

        if (more) {
#if USE_ASYNC_LDS
            stage_async((kt + 1) * BK, cur ^ 1);    // async fill of the free buffer
#else
            ldreg((kt + 1) * BK);                   // vmem in flight during WMMAs
#endif
            if (kt + 2 < kTiles) {                  // warm L2/WGP$ one more ahead
                __builtin_prefetch(aSrc + (kt + 2) * BK, 0, 1);
                __builtin_prefetch(bSrc + (kt + 2) * BK, 0, 1);
            }
        }

        // ---- fragment loads (16-bit A/B layout: lanes 0-15 K{0-7,16-23},
        //      lanes 16-31 K{8-15,24-31}; two 16B contiguous LDS reads/frag)
        v16bf afrag[4], bfrag[2];
        #pragma unroll
        for (int mi = 0; mi < 4; ++mi) {
            const bf16_t* p = &As[cur][wm * 64 + mi * 16 + r][kh];
            v8bf lo = *(const v8bf*)p;
            v8bf hi = *(const v8bf*)(p + 16);
            afrag[mi] = __builtin_shufflevector(lo, hi, 0,1,2,3,4,5,6,7,8,9,10,11,12,13,14,15);
        }
        #pragma unroll
        for (int ni = 0; ni < 2; ++ni) {
            const bf16_t* p = &Bs[cur][wn * 32 + ni * 16 + r][kh];
            v8bf lo = *(const v8bf*)p;
            v8bf hi = *(const v8bf*)(p + 16);
            bfrag[ni] = __builtin_shufflevector(lo, hi, 0,1,2,3,4,5,6,7,8,9,10,11,12,13,14,15);
        }

        // ---- 8 WMMAs per K-step per wave
        #pragma unroll
        for (int mi = 0; mi < 4; ++mi)
            #pragma unroll
            for (int ni = 0; ni < 2; ++ni)
                acc[mi][ni] = __builtin_amdgcn_wmma_f32_16x16x32_bf16(
                    false, afrag[mi], false, bfrag[ni],
                    (short)0, acc[mi][ni], false, false);

        if (more) {
#if USE_ASYNC_LDS
            wait_async0();            // my fills of buf nxt complete
            __syncthreads();          // everyone's fills visible; reads of cur done
#else
            __syncthreads();          // everyone done reading both buffers' prior data
            streg(cur ^ 1);
            __syncthreads();          // next tile visible before reads
#endif
        }
    }

    // ---- epilogue.  C frag layout: VGPR rr, lane L -> m = rr + 8*(L/16), n = L%16
    const int r8 = (lane >> 4) * 8;
    const int cn = lane & 15;
    float bv[2] = {0.f, 0.f};
    if (HAS_BIAS) {
        bv[0] = bias[n0 + wn * 32 + cn];
        bv[1] = bias[n0 + wn * 32 + 16 + cn];
    }
    #pragma unroll
    for (int mi = 0; mi < 4; ++mi) {
        #pragma unroll
        for (int ni = 0; ni < 2; ++ni) {
            #pragma unroll
            for (int rr = 0; rr < 8; ++rr) {
                float v = acc[mi][ni][rr];
                const int m = m0 + wm * 64 + mi * 16 + r8 + rr;
                const int n = n0 + wn * 32 + ni * 16 + cn;
                if (HAS_BIAS) v += bv[ni];
                if constexpr (EPI == EPI_SCORES) {
                    v *= scale;
                    if (n > m) v = -1e30f;           // causal mask
                }
                if constexpr (EPI == EPI_GELU)
                    v = 0.5f * v * (1.0f + erff(v * 0.70710678118654752f));
                const long oidx = OUT_TRANS ? ((long)n * ldOut + m)
                                            : ((long)m * ldOut + n);
                if constexpr (EPI == EPI_F32 || EPI == EPI_SCORES) {
                    ((float*)Out + (long)bb * oBS)[oidx] = v;
                } else {
                    ((bf16_t*)Out + (long)bb * oBS)[oidx] = to_bf16(v);
                }
            }
        }
    }
}

// ---------------------------------------------------------------------------
// h[b,s,d] = x[s,b,d] + PE(s,d)   (sinusoidal), converted to bf16, [B,S,D]
// ---------------------------------------------------------------------------
__global__ __launch_bounds__(256)
void add_pe_cvt(const float* __restrict__ x, bf16_t* __restrict__ hb)
{
    const long idx = (long)blockIdx.x * blockDim.x + threadIdx.x;
    const long total = (long)S_ * B_ * (D_ / 2);
    if (idx >= total) return;
    const int  j  = (int)(idx % (D_ / 2));
    const long sb = idx / (D_ / 2);
    const int  b  = (int)(sb % B_);
    const int  s  = (int)(sb / B_);
    const float divf = __expf((float)(2 * j) * (-9.21034037197618f / (float)D_)); // -ln(1e4)/D
    const float ang  = (float)s * divf;
    const float pe0  = __sinf(ang);
    const float pe1  = __cosf(ang);
    const float* xp = x + ((long)s * B_ + b) * D_ + 2 * j;
    bf16_t* hp = hb + ((long)b * S_ + s) * D_ + 2 * j;
    hp[0] = to_bf16(xp[0] + pe0);
    hp[1] = to_bf16(xp[1] + pe1);
}

// ---------------------------------------------------------------------------
// fp32 [K,N] -> bf16 transposed [N,K]  (one-time weight repack)
// ---------------------------------------------------------------------------
__global__ __launch_bounds__(256)
void cvt_transpose_bf16(const float* __restrict__ in, bf16_t* __restrict__ outT,
                        int K, int N)
{
    const long idx = (long)blockIdx.x * blockDim.x + threadIdx.x;
    const long total = (long)K * N;
    if (idx >= total) return;
    const int n = (int)(idx % N);
    const int k = (int)(idx / N);
    outT[(long)n * K + k] = to_bf16(in[idx]);
}

// ---------------------------------------------------------------------------
// causal row softmax: probs[b,s,t] = exp(sc-m)/l for t<=s else 0 (bf16 out)
// ---------------------------------------------------------------------------
__global__ __launch_bounds__(256)
void softmax_causal(const float* __restrict__ scores, bf16_t* __restrict__ probs)
{
    const int s = blockIdx.x, b = blockIdx.z;
    const float* row  = scores + ((long)b * S_ + s) * (long)S_;
    bf16_t*      prow = probs  + ((long)b * S_ + s) * (long)S_;
    const int n = s + 1;

    __shared__ float red[256];
    float mx = -1e30f;
    for (int t = threadIdx.x; t < n; t += 256) mx = fmaxf(mx, row[t]);
    red[threadIdx.x] = mx;
    __syncthreads();
    for (int w = 128; w > 0; w >>= 1) {
        if (threadIdx.x < w) red[threadIdx.x] = fmaxf(red[threadIdx.x], red[threadIdx.x + w]);
        __syncthreads();
    }
    mx = red[0];
    __syncthreads();

    float sum = 0.f;
    for (int t = threadIdx.x; t < n; t += 256) sum += __expf(row[t] - mx);
    red[threadIdx.x] = sum;
    __syncthreads();
    for (int w = 128; w > 0; w >>= 1) {
        if (threadIdx.x < w) red[threadIdx.x] += red[threadIdx.x + w];
        __syncthreads();
    }
    const float inv = 1.f / red[0];

    for (int t = threadIdx.x; t < S_; t += 256)
        prow[t] = (t < n) ? to_bf16(__expf(row[t] - mx) * inv) : to_bf16(0.f);
}

// ---------------------------------------------------------------------------
extern "C" void kernel_launch(void* const* d_in, const int* in_sizes, int n_in,
                              void* d_out, int out_size, void* d_ws, size_t ws_size,
                              hipStream_t stream)
{
    const float* x  = (const float*)d_in[0];
    const float* Wq = (const float*)d_in[1];  const float* bq = (const float*)d_in[2];
    const float* Wk = (const float*)d_in[3];  const float* bk = (const float*)d_in[4];
    const float* Wv = (const float*)d_in[5];  const float* bv = (const float*)d_in[6];
    const float* W1 = (const float*)d_in[7];  const float* b1 = (const float*)d_in[8];
    const float* W2 = (const float*)d_in[9];  const float* b2 = (const float*)d_in[10];
    (void)in_sizes; (void)n_in; (void)out_size; (void)ws_size;

    char* ws = (char*)d_ws;
    const size_t MBy = (size_t)1 << 20;
    // workspace layout (with reuse; peak ~182 MB)
    bf16_t* hb   = (bf16_t*)(ws + 0 * MBy);     // 16 MB  [B,S,D] bf16
    bf16_t* Wqt  = (bf16_t*)(ws + 16 * MBy);    //  2 MB  [A,D]
    bf16_t* Wkt  = (bf16_t*)(ws + 18 * MBy);    //  2 MB  [A,D]
    bf16_t* Wvt  = (bf16_t*)(ws + 20 * MBy);    //  2 MB  [A,D]
    bf16_t* W1t  = (bf16_t*)(ws + 22 * MBy);    //  8 MB  [F,A]
    bf16_t* W2t  = (bf16_t*)(ws + 30 * MBy);    //  8 MB  [O,F]
    bf16_t* Qb   = (bf16_t*)(ws + 38 * MBy);    // 16 MB  [B,S,A]
    bf16_t* Kb   = (bf16_t*)(ws + 54 * MBy);    // 16 MB  [B,S,A]
    bf16_t* Vt   = (bf16_t*)(ws + 70 * MBy);    // 16 MB  [B,A,S]  (V transposed)
    float*  sc   = (float*) (ws + 86 * MBy);    // 64 MB  [B,S,S] f32 scores
    bf16_t* pr   = (bf16_t*)(ws + 150 * MBy);   // 32 MB  [B,S,S] bf16 probs
    bf16_t* attn = (bf16_t*)(ws + 38 * MBy);    // 16 MB  [S,B,A] (reuses Q)
    bf16_t* h1   = (bf16_t*)(ws + 86 * MBy);    // 64 MB  [S*B,F] (reuses scores)

    auto cgrid = [](long nv) { return dim3((unsigned)((nv + 255) / 256)); };

    // 1) weight repack (transposed bf16) + PE fuse
    cvt_transpose_bf16<<<cgrid((long)D_ * A_), 256, 0, stream>>>(Wq, Wqt, D_, A_);
    cvt_transpose_bf16<<<cgrid((long)D_ * A_), 256, 0, stream>>>(Wk, Wkt, D_, A_);
    cvt_transpose_bf16<<<cgrid((long)D_ * A_), 256, 0, stream>>>(Wv, Wvt, D_, A_);
    cvt_transpose_bf16<<<cgrid((long)A_ * F_), 256, 0, stream>>>(W1, W1t, A_, F_);
    cvt_transpose_bf16<<<cgrid((long)F_ * O_), 256, 0, stream>>>(W2, W2t, F_, O_);
    add_pe_cvt<<<cgrid((long)S_ * B_ * (D_ / 2)), 256, 0, stream>>>(x, hb);

    const int MQ = B_ * S_;  // 8192 rows for projections / FFN

    // 2) Q, K projections (bf16 out, row-major)
    gemm_wmma<EPI_BF16, true, false, false, false>
        <<<dim3(A_ / BN, MQ / BM, 1), 256, 0, stream>>>(
            hb, D_, Wqt, D_, Qb, A_, D_, bq, 1.f, 0, 0, 0);
    gemm_wmma<EPI_BF16, true, false, false, false>
        <<<dim3(A_ / BN, MQ / BM, 1), 256, 0, stream>>>(
            hb, D_, Wkt, D_, Kb, A_, D_, bk, 1.f, 0, 0, 0);
    // V projection writes V^T [B,A,S] directly (batched over z)
    gemm_wmma<EPI_BF16, true, false, false, true>
        <<<dim3(A_ / BN, S_ / BM, B_), 256, 0, stream>>>(
            hb, D_, Wvt, D_, Vt, S_, D_, bv, 1.f,
            (long)S_ * D_, 0, (long)A_ * S_);

    // 3) scores = Q @ K^T / sqrt(A), upper-triangle tiles skipped, f32 out
    gemm_wmma<EPI_SCORES, false, true, false, false>
        <<<dim3(S_ / BN, S_ / BM, B_), 256, 0, stream>>>(
            Qb, A_, Kb, A_, sc, S_, A_, nullptr, 0.03125f,
            (long)S_ * A_, (long)S_ * A_, (long)S_ * S_);

    // 4) causal softmax -> bf16 probs
    softmax_causal<<<dim3(S_, 1, B_), 256, 0, stream>>>(sc, pr);

    // 5) attn = P @ V (B operand = V^T), K-loop limited to the causal band;
    //    epilogue scatters to [S,B,A]
    gemm_wmma<EPI_BF16, false, false, true, false>
        <<<dim3(A_ / BN, S_ / BM, B_), 256, 0, stream>>>(
            pr, S_, Vt, S_, attn, B_ * A_, S_, nullptr, 1.f,
            (long)S_ * S_, (long)A_ * S_, (long)A_);

    // 6) FFN1: gelu(attn @ W1 + b1), bf16 out
    gemm_wmma<EPI_GELU, true, false, false, false>
        <<<dim3(F_ / BN, MQ / BM, 1), 256, 0, stream>>>(
            attn, A_, W1t, A_, h1, F_, A_, b1, 1.f, 0, 0, 0);

    // 7) FFN2: h1 @ W2 + b2 -> f32 d_out [S,B,O]
    gemm_wmma<EPI_F32, true, false, false, false>
        <<<dim3(O_ / BN, MQ / BM, 1), 256, 0, stream>>>(
            h1, F_, W2t, F_, (float*)d_out, O_, F_, b2, 1.f, 0, 0, 0);
}
